// MultiHeadAttention_23390391894406
// MI455X (gfx1250) — matrix operand order repaired and checked by reading, not compiled, not run on hardware
//
#include <hip/hip_runtime.h>
#include <hip/hip_bf16.h>

// MHA for B=4, S=2048, D=1024, H=16, HD=64 on gfx1250 (wave32, WMMA bf16).

typedef __attribute__((ext_vector_type(16))) __bf16 v16bf;
typedef __attribute__((ext_vector_type(8)))  __bf16 v8bf;
typedef __attribute__((ext_vector_type(4)))  __bf16 v4bf;
typedef __attribute__((ext_vector_type(8)))  float  v8f;
typedef __attribute__((ext_vector_type(4)))  unsigned int u32x4;
typedef __attribute__((ext_vector_type(8)))  int i32x8;
typedef __attribute__((ext_vector_type(4)))  int i32x4;

constexpr int Bc = 4, Sc = 2048, Dc = 1024, Hc = 16, HDc = 64;

// ---- CDNA5 feature detection (safe: falls back to plain loads) -------------
#if __has_builtin(__builtin_amdgcn_global_load_async_to_lds_b128) && \
    __has_builtin(__builtin_amdgcn_s_wait_asynccnt)
#define ATT_ASYNC 1
#else
#define ATT_ASYNC 0
#endif

#if __has_builtin(__builtin_amdgcn_tensor_load_to_lds) && \
    __has_builtin(__builtin_amdgcn_s_wait_tensorcnt)
#define ATT_TDM 1
#else
#define ATT_TDM 0
#endif

#if __has_include(<hip/amd_detail/amd_gfx1250_TDM.h>)
#define TDM_SIX_ARGS 1   // therock/clang-23 arity
#else
#define TDM_SIX_ARGS 0   // ROCm 7.2 clang-22 arity
#endif

#if __has_builtin(__builtin_amdgcn_permlane16)
#define HAS_PERMLANE16 1
#else
#define HAS_PERMLANE16 0
#endif

__device__ inline v8f v8f_zero() {
    v8f z;
#pragma unroll
    for (int i = 0; i < 8; ++i) z[i] = 0.0f;
    return z;
}

__device__ inline v8f wmma_bf16(v16bf a, v16bf b, v8f c) {
    // D(f32,16x16) = A(bf16,16x32) * B(bf16,32x16) + C
    return __builtin_amdgcn_wmma_f32_16x16x32_bf16(
        /*neg_a=*/false, a, /*neg_b=*/false, b,
        /*c_mod=*/(short)0, c, /*reuse_a=*/false, /*reuse_b=*/false);
}

// A-fragment (16x32 bf16): lane L(<16): row=L, elems K={e0..e0+7, e0+16..e0+23},
// e0 = 8*(lane/16). Source row-major LDS tile with given element pitch.
__device__ inline v16bf load_a_frag(const __bf16* rowbase, int pitch, int l16, int half, int kbase) {
    const __bf16* p = rowbase + l16 * pitch + kbase + half * 8;
    union { v16bf v; v8bf h[2]; } u;
    u.h[0] = *(const v8bf*)(p);
    u.h[1] = *(const v8bf*)(p + 16);
    return u.v;
}

// B-fragment (32x16 bf16) from LDS stored transposed as [N][K]: lane col=L%16,
// elems K={e0..e0+7, e0+16..e0+23}, e0 = 8*(lane/16) (same order as A).
__device__ inline v16bf load_b_frag(const __bf16* nbase, int pitch, int l16, int half, int kbase) {
    const __bf16* p = nbase + l16 * pitch + kbase + half * 8;
    union { v16bf v; v8bf h[2]; } u;
    u.h[0] = *(const v8bf*)(p);
    u.h[1] = *(const v8bf*)(p + 16);
    return u.v;
}

// 16-lane XOR-butterfly reductions (rows of 16 match the C-matrix lane halves).
__device__ inline float pl16(float v, unsigned lo, unsigned hi) {
#if HAS_PERMLANE16
    int r = __builtin_amdgcn_permlane16(__float_as_int(v), __float_as_int(v),
                                        (int)lo, (int)hi, false, false);
    return __int_as_float(r);
#else
    (void)lo; (void)hi;
    return v;
#endif
}

__device__ inline float row_max16(float v) {
#if HAS_PERMLANE16
    v = fmaxf(v, pl16(v, 0x67452301u, 0xEFCDAB89u));  // xor 1
    v = fmaxf(v, pl16(v, 0x45670123u, 0xCDEF89ABu));  // xor 2
    v = fmaxf(v, pl16(v, 0x01234567u, 0x89ABCDEFu));  // xor 4
    v = fmaxf(v, pl16(v, 0xFEDCBA98u, 0x76543210u));  // xor 8
#else
#pragma unroll
    for (int off = 1; off < 16; off <<= 1) v = fmaxf(v, __shfl_xor(v, off, 32));
#endif
    return v;
}

__device__ inline float row_sum16(float v) {
#if HAS_PERMLANE16
    v += pl16(v, 0x67452301u, 0xEFCDAB89u);
    v += pl16(v, 0x45670123u, 0xCDEF89ABu);
    v += pl16(v, 0x01234567u, 0x89ABCDEFu);
    v += pl16(v, 0xFEDCBA98u, 0x76543210u);
#else
#pragma unroll
    for (int off = 1; off < 16; off <<= 1) v += __shfl_xor(v, off, 32);
#endif
    return v;
}

#if ATT_ASYNC || ATT_TDM
__device__ inline unsigned lds_off(const void* p) {
    return (unsigned)(size_t)(__attribute__((address_space(3))) const void*)p;
}
#endif

#if ATT_ASYNC
// Async 16-byte copy global -> LDS (tracked by ASYNCcnt).
// Builtin prototype: (int4 addrspace(1)* src, int4 addrspace(3)* dst, imm offset, imm cpol)
__device__ inline void async_copy_b128(const __bf16* g, __bf16* l) {
    __builtin_amdgcn_global_load_async_to_lds_b128(
        (__attribute__((address_space(1))) i32x4*)(__attribute__((address_space(1))) void*)(void*)(__bf16*)g,
        (__attribute__((address_space(3))) i32x4*)(__attribute__((address_space(3))) void*)(void*)l,
        0, 0);
}
#endif

// ---------------------------------------------------------------------------
// GEMM: out[M,N] = (A[M,K] * W[K,N] + bias[N]) * oscale
// ABF16: A is bf16 (else f32, converted during LDS staging). W/bias f32.
// REMAP: store bf16 to [B,H,S,HD] (projections); else f32 row-major.
// 128x128x64 block tile, 8 waves (2x4), each wave 64x32 -> 4x2 accumulators.
// ---------------------------------------------------------------------------
template <bool ABF16, bool REMAP>
__global__ __launch_bounds__(256) void gemm_bias(const void* __restrict__ Ap,
                                                 const float* __restrict__ W,
                                                 const float* __restrict__ bias,
                                                 void* __restrict__ Op,
                                                 int M, int N, int K, float oscale) {
    constexpr int BM = 128, BN = 128, BK = 64, LP = BK + 8;  // pitch 72 (144B, 16B-aligned)
    __shared__ __bf16 As[BM * LP];
    __shared__ __bf16 Bs[BN * LP];  // W tile transposed: Bs[n][k]

    const int tid  = threadIdx.x;
    const int wid  = tid >> 5;
    const int lane = tid & 31;
    const int half = lane >> 4;
    const int l16  = lane & 15;
    const int wm   = wid >> 2;
    const int wn   = wid & 3;
    const int m0   = blockIdx.y * BM;
    const int n0   = blockIdx.x * BN;
    (void)M;

    v8f acc[4][2];
#pragma unroll
    for (int i = 0; i < 4; ++i)
#pragma unroll
        for (int j = 0; j < 2; ++j) acc[i][j] = v8f_zero();

    for (int kt = 0; kt < K; kt += BK) {
        __syncthreads();
        // Stage A tile (128 x 64) -> bf16 LDS
#pragma unroll
        for (int i = 0; i < 8; ++i) {
            int c  = tid + i * 256;
            int r  = c >> 4;
            int cc = (c & 15) * 4;
            v4bf pk;
            if constexpr (ABF16) {
                const __bf16* A = (const __bf16*)Ap;
                pk = *(const v4bf*)(A + (size_t)(m0 + r) * K + kt + cc);
            } else {
                const float* A = (const float*)Ap;
                float4 f = *(const float4*)(A + (size_t)(m0 + r) * K + kt + cc);
                pk[0] = (__bf16)f.x; pk[1] = (__bf16)f.y;
                pk[2] = (__bf16)f.z; pk[3] = (__bf16)f.w;
            }
            *(v4bf*)(&As[r * LP + cc]) = pk;
        }
        // Stage W tile (64 x 128), transposed into Bs[n][k]
#pragma unroll
        for (int i = 0; i < 8; ++i) {
            int c  = tid + i * 256;
            int kr = c >> 5;
            int nc = (c & 31) * 4;
            float4 f = *(const float4*)(W + (size_t)(kt + kr) * N + n0 + nc);
            Bs[(nc + 0) * LP + kr] = (__bf16)f.x;
            Bs[(nc + 1) * LP + kr] = (__bf16)f.y;
            Bs[(nc + 2) * LP + kr] = (__bf16)f.z;
            Bs[(nc + 3) * LP + kr] = (__bf16)f.w;
        }
        __syncthreads();

#pragma unroll
        for (int ks = 0; ks < 2; ++ks) {
            v16bf af[4], bf[2];
#pragma unroll
            for (int mt = 0; mt < 4; ++mt)
                af[mt] = load_a_frag(&As[(wm * 64 + mt * 16) * LP], LP, l16, half, ks * 32);
#pragma unroll
            for (int nt = 0; nt < 2; ++nt)
                bf[nt] = load_b_frag(&Bs[(wn * 32 + nt * 16) * LP], LP, l16, half, ks * 32);
#pragma unroll
            for (int mt = 0; mt < 4; ++mt)
#pragma unroll
                for (int nt = 0; nt < 2; ++nt)
                    acc[mt][nt] = wmma_bf16(af[mt], bf[nt], acc[mt][nt]);
        }
    }

    // Epilogue: C layout -> lane l16 = N; (VGPR r + 8*half) = M within 16x16 tile.
#pragma unroll
    for (int nt = 0; nt < 2; ++nt) {
        int n = n0 + wn * 32 + nt * 16 + l16;
        float bv = bias[n];
#pragma unroll
        for (int mt = 0; mt < 4; ++mt) {
#pragma unroll
            for (int r = 0; r < 8; ++r) {
                int m = m0 + wm * 64 + mt * 16 + half * 8 + r;
                float v = (acc[mt][nt][r] + bv) * oscale;
                if constexpr (REMAP) {
                    int b = m / Sc, s = m % Sc, h = n >> 6, hd = n & 63;
                    ((__bf16*)Op)[(((size_t)b * Hc + h) * Sc + s) * HDc + hd] = (__bf16)v;
                } else {
                    ((float*)Op)[(size_t)m * N + n] = v;
                }
            }
        }
    }
}

// ---------------------------------------------------------------------------
// Flash attention: grid = (S/128, B*H), 256 threads; wave owns 16 query rows.
// Q staged via async global->LDS (ASYNCcnt); K tile streamed by the Tensor
// Data Mover with LDS padding matching the pitch-72 layout (TENSORcnt);
// V transposed manually; P re-laid-out via per-wave LDS for the PV WMMA.
// ---------------------------------------------------------------------------
__global__ __launch_bounds__(256) void attention_kernel(const __bf16* __restrict__ Qw,
                                                        const __bf16* __restrict__ Kw,
                                                        const __bf16* __restrict__ Vw,
                                                        __bf16* __restrict__ Cw) {
    constexpr int KB = 64, P = 72;      // pitch 72 elems = 144B (16B aligned)
    __shared__ __bf16 Qs[128 * P];      // 128 x 64 Q block
    __shared__ __bf16 Ks[KB * P];       // K block [kk][d] == B layout for Q*K^T
    __shared__ __bf16 Vs[HDc * P];      // V^T block [hd][kk]
    __shared__ __bf16 Ps[8][16 * P];    // per-wave P 16x64 in A layout

    const int tid  = threadIdx.x;
    const int wid  = tid >> 5;
    const int lane = tid & 31;
    const int half = lane >> 4;
    const int l16  = lane & 15;

    const int bh = blockIdx.y;
    const int q0 = blockIdx.x * 128;
    const __bf16* Qp = Qw + ((size_t)bh * Sc + q0) * HDc;
    const __bf16* Kp = Kw + (size_t)bh * Sc * HDc;
    const __bf16* Vp = Vw + (size_t)bh * Sc * HDc;

    // Load Q block (128 x 64 bf16), row-major (pre-scaled by 1/sqrt(HD)).
#pragma unroll
    for (int i = 0; i < 4; ++i) {
        int c = tid + i * 256;
        int r = c >> 3, cc = (c & 7) * 8;
#if ATT_ASYNC
        async_copy_b128(Qp + (size_t)r * HDc + cc, &Qs[r * P + cc]);
#else
        *(v8bf*)(&Qs[r * P + cc]) = *(const v8bf*)(Qp + (size_t)r * HDc + cc);
#endif
    }
#if ATT_ASYNC
    __builtin_amdgcn_s_wait_asynccnt(0);
#endif

    v8f o[4];
#pragma unroll
    for (int i = 0; i < 4; ++i) o[i] = v8f_zero();
    float mrow[8], lrow[8];
#pragma unroll
    for (int r = 0; r < 8; ++r) { mrow[r] = -3.0e30f; lrow[r] = 0.0f; }

    for (int kb = 0; kb < Sc; kb += KB) {
        __syncthreads();

        // ---- K block 64x64 (raw bf16 copy into padded LDS) ----
#if ATT_TDM
        if (wid == 0) {
            unsigned long long ga = (unsigned long long)(size_t)(Kp + (size_t)kb * HDc);
            // D# group0: count=1 | lds_addr | 57b global_addr | type=2(image)
            u32x4 g0 = { 1u, lds_off(Ks), (unsigned)ga,
                         (unsigned)(ga >> 32) | 0x80000000u };
            // D# group1: data_size=2B, pad_enable, pad_interval=32dw, pad_amount=4dw;
            // tensor_dim0=64, tensor_dim1=2048, tile_dim0=64, tile_dim1=64,
            // tensor_dim0_stride=64.
            i32x8 g1 = { (int)0x07110000u,
                         (int)(64u << 16),    // tensor_dim0[15:0] @ bits 63:48
                         (int)(2048u << 16),  // dim0[31:16]=0 | tensor_dim1[15:0]
                         (int)(64u << 16),    // dim1[31:16]=0 | tile_dim0=64
                         64,                  // tile_dim1=64 | tile_dim2=0
                         64,                  // tensor_dim0_stride[31:0]
                         0, 0 };
            i32x4 gz = { 0, 0, 0, 0 };
#if TDM_SIX_ARGS
            i32x8 gz8 = { 0, 0, 0, 0, 0, 0, 0, 0 };
            __builtin_amdgcn_tensor_load_to_lds(g0, g1, gz, gz, gz8, 0);
#else
            __builtin_amdgcn_tensor_load_to_lds(g0, g1, gz, gz, 0);
#endif
        }
#else
#pragma unroll
        for (int i = 0; i < 2; ++i) {
            int c = tid + i * 256;
            int r = c >> 3, cc = (c & 7) * 8;
#if ATT_ASYNC
            async_copy_b128(Kp + (size_t)(kb + r) * HDc + cc, &Ks[r * P + cc]);
#else
            *(v8bf*)(&Ks[r * P + cc]) = *(const v8bf*)(Kp + (size_t)(kb + r) * HDc + cc);
#endif
        }
#endif

        // ---- V block 64x64, transposed into Vs[hd][kk] (B layout for P*V) ----
#pragma unroll
        for (int i = 0; i < 2; ++i) {
            int c = tid + i * 256;
            int r = c >> 3, h0 = (c & 7) * 8;
            v8bf v = *(const v8bf*)(Vp + (size_t)(kb + r) * HDc + h0);
#pragma unroll
            for (int j = 0; j < 8; ++j) Vs[(h0 + j) * P + r] = v[j];
        }

#if ATT_TDM
        if (wid == 0) __builtin_amdgcn_s_wait_tensorcnt(0);
#elif ATT_ASYNC
        __builtin_amdgcn_s_wait_asynccnt(0);
#endif
        __syncthreads();

        // ---- S = Q * K^T : 16 rows x 64 cols per wave, 2 k-steps ----
        v8f st[4];
#pragma unroll
        for (int i = 0; i < 4; ++i) st[i] = v8f_zero();
#pragma unroll
        for (int ks = 0; ks < 2; ++ks) {
            v16bf a = load_a_frag(&Qs[wid * 16 * P], P, l16, half, ks * 32);
#pragma unroll
            for (int nt = 0; nt < 4; ++nt) {
                v16bf b = load_b_frag(&Ks[nt * 16 * P], P, l16, half, ks * 32);
                st[nt] = wmma_bf16(a, b, st[nt]);
            }
        }

        // ---- Online softmax: element st[t][r] is (row r+8*half, col t*16+l16) ----
#pragma unroll
        for (int r = 0; r < 8; ++r) {
            float cur = fmaxf(fmaxf(st[0][r], st[1][r]), fmaxf(st[2][r], st[3][r]));
            cur = row_max16(cur);
            float nm = fmaxf(mrow[r], cur);
            float al = __expf(mrow[r] - nm);
            mrow[r] = nm;
            float rs = 0.0f;
#pragma unroll
            for (int t = 0; t < 4; ++t) {
                float p = __expf(st[t][r] - nm);
                st[t][r] = p;
                rs += p;
            }
            rs = row_sum16(rs);
            lrow[r] = lrow[r] * al + rs;
#pragma unroll
            for (int ot = 0; ot < 4; ++ot) o[ot][r] *= al;
        }

        // ---- Re-layout P (C layout -> A layout) through per-wave LDS ----
        __bf16* Pw = Ps[wid];
#pragma unroll
        for (int t = 0; t < 4; ++t)
#pragma unroll
            for (int r = 0; r < 8; ++r)
                Pw[(half * 8 + r) * P + t * 16 + l16] = (__bf16)st[t][r];
        __syncthreads();

        // ---- O += P * V : 2 k-steps x 4 hd-tiles ----
#pragma unroll
        for (int ks = 0; ks < 2; ++ks) {
            v16bf a = load_a_frag(Pw, P, l16, half, ks * 32);
#pragma unroll
            for (int ot = 0; ot < 4; ++ot) {
                v16bf b = load_b_frag(&Vs[ot * 16 * P], P, l16, half, ks * 32);
                o[ot] = wmma_bf16(a, b, o[ot]);
            }
        }
    }

    // Normalize and store ctx as bf16 [B,S,D].
    const int b = bh / Hc, h = bh % Hc;
#pragma unroll
    for (int r = 0; r < 8; ++r) {
        float inv = 1.0f / lrow[r];
        int q = q0 + wid * 16 + half * 8 + r;
#pragma unroll
        for (int ot = 0; ot < 4; ++ot) {
            int hd = ot * 16 + l16;
            Cw[((size_t)b * Sc + q) * Dc + h * HDc + hd] = (__bf16)(o[ot][r] * inv);
        }
    }
}

extern "C" void kernel_launch(void* const* d_in, const int* in_sizes, int n_in,
                              void* d_out, int out_size, void* d_ws, size_t ws_size,
                              hipStream_t stream) {
    (void)in_sizes; (void)n_in; (void)out_size; (void)ws_size;
    const float* query = (const float*)d_in[0];
    const float* key   = (const float*)d_in[1];
    const float* value = (const float*)d_in[2];
    const float* Wq = (const float*)d_in[3];  const float* bq = (const float*)d_in[4];
    const float* Wk = (const float*)d_in[5];  const float* bk = (const float*)d_in[6];
    const float* Wv = (const float*)d_in[7];  const float* bv = (const float*)d_in[8];
    const float* Wo = (const float*)d_in[9];  const float* bo = (const float*)d_in[10];

    const size_t elems = (size_t)Bc * Sc * Dc;   // 8M elements
    __bf16* Qw = (__bf16*)d_ws;                  // 16 MB each (bf16)
    __bf16* Kw = Qw + elems;
    __bf16* Vw = Kw + elems;
    __bf16* Cw = Vw + elems;                     // 64 MB workspace total

    dim3 blk(256);
    dim3 gproj(Dc / 128, (Bc * Sc) / 128);       // (8, 64)

    // Projections: f32 in, bf16 [B,H,S,HD] out. Q pre-scaled by 1/sqrt(HD).
    gemm_bias<false, true><<<gproj, blk, 0, stream>>>(query, Wq, bq, Qw, Bc * Sc, Dc, Dc, 0.125f);
    gemm_bias<false, true><<<gproj, blk, 0, stream>>>(key,   Wk, bk, Kw, Bc * Sc, Dc, Dc, 1.0f);
    gemm_bias<false, true><<<gproj, blk, 0, stream>>>(value, Wv, bv, Vw, Bc * Sc, Dc, Dc, 1.0f);

    // Attention.
    dim3 gatt(Sc / 128, Bc * Hc);                // (16, 64)
    attention_kernel<<<gatt, blk, 0, stream>>>(Qw, Kw, Vw, Cw);

    // Output projection: bf16 ctx in, f32 out.
    gemm_bias<true, false><<<gproj, blk, 0, stream>>>(Cw, Wo, bo, d_out, Bc * Sc, Dc, Dc, 1.0f);
}
